// moleCore_49615462203810
// MI455X (gfx1250) — matrix operand order repaired and checked by reading, not compiled, not run on hardware
//
#include <hip/hip_runtime.h>
#include <math.h>

// ---------------- problem constants ----------------
constexpr int T = 16384, D = 1024, F = 4096, E = 16, K = 2, C = 2560;

typedef __attribute__((ext_vector_type(16))) __bf16 v16bf;
typedef __attribute__((ext_vector_type(8)))  float  v8f;

union FragBF { v16bf v; uint4 q[2]; };

// ---------------- CDNA5 async global->LDS helpers ----------------
__device__ __forceinline__ void async_b128(unsigned lds_off, const __bf16* g) {
  // GLOBAL_LOAD_ASYNC_TO_LDS_B128: VDST = LDS byte address, VADDR = 64-bit global addr
  asm volatile("global_load_async_to_lds_b128 %0, %1, off"
               :: "v"(lds_off), "v"(g) : "memory");
}
__device__ __forceinline__ void wait_async0() {
  asm volatile("s_wait_asynccnt 0" ::: "memory");
}

// ---------------- kernel 1: gating + top-2 ----------------
__global__ __launch_bounds__(256) void gate_topk_kernel(
    const float* __restrict__ x, const float* __restrict__ gw,
    int* __restrict__ eid, float* __restrict__ topv) {
  int t = blockIdx.x * blockDim.x + threadIdx.x;
  if (t >= T) return;
  float acc[E];
#pragma unroll
  for (int e = 0; e < E; ++e) acc[e] = 0.f;
  const float4* x4 = (const float4*)(x + (size_t)t * D);
  for (int d4 = 0; d4 < D / 4; ++d4) {
    float4 xv = x4[d4];
    const float* g0 = gw + (size_t)(d4 * 4) * E;
#pragma unroll
    for (int e = 0; e < E; ++e) {
      acc[e] += xv.x * g0[e] + xv.y * g0[E + e] + xv.z * g0[2 * E + e] + xv.w * g0[3 * E + e];
    }
  }
  int i1 = 0; float m1 = acc[0];
#pragma unroll
  for (int e = 1; e < E; ++e) if (acc[e] > m1) { m1 = acc[e]; i1 = e; }
  int i2 = (i1 == 0) ? 1 : 0; float m2 = acc[i2];
#pragma unroll
  for (int e = 0; e < E; ++e) if (e != i1 && acc[e] > m2) { m2 = acc[e]; i2 = e; }
  float w0 = 1.f / (1.f + __expf(m2 - m1));
  eid[t * 2] = i1; eid[t * 2 + 1] = i2;
  topv[t * 2] = w0; topv[t * 2 + 1] = 1.f - w0;
}

// ---------------- kernel 2: ordered dispatch (token-major prefix scan) ----------------
__global__ __launch_bounds__(512) void dispatch_kernel(
    const int* __restrict__ eid, int* __restrict__ tok_slot,
    int* __restrict__ slot_token) {
  constexpr int NT = 512, PER = (T * K) / NT;   // 64 flat entries per thread
  __shared__ int cnts[NT * E];                  // 32 KB
  int j = threadIdx.x;
  for (int s = j; s < E * C; s += NT) slot_token[s] = -1;
  int local[E];
#pragma unroll
  for (int e = 0; e < E; ++e) local[e] = 0;
  int base = j * PER;
  for (int i = 0; i < PER; ++i) local[eid[base + i]]++;
#pragma unroll
  for (int e = 0; e < E; ++e) cnts[j * E + e] = local[e];
  __syncthreads();
  if (j < E) {
    int run = 0;
    for (int jj = 0; jj < NT; ++jj) { int v = cnts[jj * E + j]; cnts[jj * E + j] = run; run += v; }
  }
  __syncthreads();
#pragma unroll
  for (int e = 0; e < E; ++e) local[e] = cnts[j * E + e];
  for (int i = 0; i < PER; ++i) {
    int idx = base + i;
    int e = eid[idx];
    int loc = local[e]++;
    if (loc < C) {
      int s = e * C + loc;
      tok_slot[idx] = s;
      slot_token[s] = idx / K;
    } else {
      tok_slot[idx] = -1;
    }
  }
}

// ---------------- kernel 3: x -> bf16 ----------------
__global__ __launch_bounds__(256) void cvt_x_kernel(const float* __restrict__ x,
                                                    __bf16* __restrict__ xb) {
  size_t i = ((size_t)blockIdx.x * blockDim.x + threadIdx.x) * 4;
  float4 v = *(const float4*)(x + i);
  __bf16* o = xb + i;
  o[0] = (__bf16)v.x; o[1] = (__bf16)v.y; o[2] = (__bf16)v.z; o[3] = (__bf16)v.w;
}

// ---------------- kernel 4: transpose-convert weights to N-major bf16 ----------------
__global__ __launch_bounds__(256) void tconv_kernel(const float* __restrict__ src,
                                                    __bf16* __restrict__ dst,
                                                    int R, int Cc) {
  __shared__ float tile[32][33];
  int e = blockIdx.z;
  int r0 = blockIdx.y * 32, c0 = blockIdx.x * 32;
  const float* s = src + (size_t)e * R * Cc;
  __bf16* d = dst + (size_t)e * R * Cc;
  int tx = threadIdx.x, ty = threadIdx.y;
#pragma unroll
  for (int i = 0; i < 4; ++i)
    tile[ty + 8 * i][tx] = s[(size_t)(r0 + ty + 8 * i) * Cc + c0 + tx];
  __syncthreads();
#pragma unroll
  for (int i = 0; i < 4; ++i)
    d[(size_t)(c0 + ty + 8 * i) * R + r0 + tx] = (__bf16)tile[tx][ty + 8 * i];
}

// ---------------- WMMA fragment loads (contiguous k = 16h + s, same order for A and B) ----------------
constexpr int LDT = 72;  // padded LDS row stride (elements): 144B, 16B-aligned, bank-safe

__device__ __forceinline__ v16bf load_fragA(const __bf16* As, int m, int kc, int lane) {
  // A 16x32 bf16: lane<16 -> K 0..15, lane>=16 -> K 16..31 (contig 32B)
  int ah = ((lane >> 4) & 1) * 16;
  const __bf16* p = As + m * LDT + kc + ah;
  FragBF f;
  f.q[0] = *(const uint4*)p;
  f.q[1] = *(const uint4*)(p + 8);
  return f.v;
}
__device__ __forceinline__ v16bf load_fragB(const __bf16* Bs, int n, int kc, int lane) {
  // B 32x16 bf16 (N-major in LDS): lane<16 -> K 0..15, lane>=16 -> K 16..31 (contig 32B)
  int bh = ((lane >> 4) & 1) * 16;
  const __bf16* p = Bs + n * LDT + kc + bh;
  FragBF f;
  f.q[0] = *(const uint4*)p;
  f.q[1] = *(const uint4*)(p + 8);
  return f.v;
}

// ---------------- kernel 5: expert GEMM1 (gather + x@w1 + b1 + gelu -> h bf16) ----------------
__global__ __launch_bounds__(256) void gemm1_kernel(
    const __bf16* __restrict__ xb, const __bf16* __restrict__ w1t,
    const float* __restrict__ b1, const int* __restrict__ slot_token,
    __bf16* __restrict__ h) {
  constexpr int BM = 128, BN = 128, BK = 64;
  constexpr int BUFE = (BM + BN) * LDT;          // elements per double-buffer slab
  __shared__ __bf16 smem[2 * BUFE];              // 72 KB
  const int e = blockIdx.z;
  const int cBase = blockIdx.y * BM;
  const int nBase = blockIdx.x * BN;
  const int tid = threadIdx.x;
  const int lane = tid & 31, wave = tid >> 5;
  const int wm = wave & 3, wn = wave >> 2;
  const int rowL = tid >> 1, hlf = tid & 1;
  const int lm = lane & 15;

  // Rows of unassigned slots are never gathered downstream -> clamp to row 0,
  // avoiding a divergent exec-mask path in the hot loop.
  int tok = slot_token[e * C + cBase + rowL];
  if (tok < 0) tok = 0;
  const __bf16* aRow = xb + (size_t)tok * D + hlf * 32;
  const __bf16* bRow = w1t + ((size_t)e * F + nBase + rowL) * D + hlf * 32;

  const unsigned smemBase = (unsigned)(size_t)smem;          // LDS byte offset
  const unsigned aOff = smemBase + (unsigned)((rowL * LDT + hlf * 32) * 2);
  const unsigned bOff = aOff + (unsigned)(BM * LDT * 2);
  const unsigned bufBytes = BUFE * 2;

  auto issue = [&](int buf, int kb) {
    const __bf16* ap = aRow + kb * BK;
    const __bf16* bp = bRow + kb * BK;
    unsigned ab = aOff + (unsigned)buf * bufBytes;
    unsigned bb = bOff + (unsigned)buf * bufBytes;
#pragma unroll
    for (int c = 0; c < 4; ++c) {
      async_b128(ab + 16 * c, ap + 8 * c);
      async_b128(bb + 16 * c, bp + 8 * c);
    }
  };

  v8f acc[2][4];
  const v8f vzero = {0.f, 0.f, 0.f, 0.f, 0.f, 0.f, 0.f, 0.f};
#pragma unroll
  for (int mi = 0; mi < 2; ++mi)
#pragma unroll
    for (int ni = 0; ni < 4; ++ni) acc[mi][ni] = vzero;

  constexpr int KB = D / BK;  // 16
  issue(0, 0);
  for (int kb = 0; kb < KB; ++kb) {
    const int cur = kb & 1;
    wait_async0();            // my async writes for buf[cur] are done
    __syncthreads();          // everyone's writes are done, prior reads retired
    if (kb + 1 < KB) issue(cur ^ 1, kb + 1);
    const __bf16* Acur = smem + cur * BUFE;
    const __bf16* Bcur = Acur + BM * LDT;
#pragma unroll
    for (int ks = 0; ks < BK; ks += 32) {
      v16bf afr[2], bfr[4];
#pragma unroll
      for (int mi = 0; mi < 2; ++mi) afr[mi] = load_fragA(Acur, wm * 32 + mi * 16 + lm, ks, lane);
#pragma unroll
      for (int ni = 0; ni < 4; ++ni) bfr[ni] = load_fragB(Bcur, wn * 64 + ni * 16 + lm, ks, lane);
#pragma unroll
      for (int mi = 0; mi < 2; ++mi)
#pragma unroll
        for (int ni = 0; ni < 4; ++ni)
          acc[mi][ni] = __builtin_amdgcn_wmma_f32_16x16x32_bf16(
              false, afr[mi], false, bfr[ni], (short)0, acc[mi][ni], false, false);
    }
  }

  const int rOff = (lane < 16) ? 0 : 8;
#pragma unroll
  for (int ni = 0; ni < 4; ++ni) {
    int col = nBase + wn * 64 + ni * 16 + lm;
    float bias = b1[e * F + col];
#pragma unroll
    for (int mi = 0; mi < 2; ++mi) {
#pragma unroll
      for (int r = 0; r < 8; ++r) {
        int row = cBase + wm * 32 + mi * 16 + r + rOff;
        float u = acc[mi][ni][r] + bias;
        float tt = tanhf(0.7978845608028654f * (u + 0.044715f * u * u * u));
        h[((size_t)e * C + row) * F + col] = (__bf16)(0.5f * u * (1.0f + tt));
      }
    }
  }
}

// ---------------- kernel 6: expert GEMM2 (h@w2 + b2 -> ye f32) ----------------
__global__ __launch_bounds__(256) void gemm2_kernel(
    const __bf16* __restrict__ h, const __bf16* __restrict__ w2t,
    const float* __restrict__ b2, float* __restrict__ ye) {
  constexpr int BM = 128, BN = 128, BK = 64;
  constexpr int BUFE = (BM + BN) * LDT;
  __shared__ __bf16 smem[2 * BUFE];
  const int e = blockIdx.z;
  const int cBase = blockIdx.y * BM;
  const int nBase = blockIdx.x * BN;
  const int tid = threadIdx.x;
  const int lane = tid & 31, wave = tid >> 5;
  const int wm = wave & 3, wn = wave >> 2;
  const int rowL = tid >> 1, hlf = tid & 1;
  const int lm = lane & 15;

  const __bf16* aRow = h + ((size_t)e * C + cBase + rowL) * F + hlf * 32;
  const __bf16* bRow = w2t + ((size_t)e * D + nBase + rowL) * F + hlf * 32;

  const unsigned smemBase = (unsigned)(size_t)smem;
  const unsigned aOff = smemBase + (unsigned)((rowL * LDT + hlf * 32) * 2);
  const unsigned bOff = aOff + (unsigned)(BM * LDT * 2);
  const unsigned bufBytes = BUFE * 2;

  auto issue = [&](int buf, int kb) {
    const __bf16* ap = aRow + kb * BK;
    const __bf16* bp = bRow + kb * BK;
    unsigned ab = aOff + (unsigned)buf * bufBytes;
    unsigned bb = bOff + (unsigned)buf * bufBytes;
#pragma unroll
    for (int c = 0; c < 4; ++c) {
      async_b128(ab + 16 * c, ap + 8 * c);
      async_b128(bb + 16 * c, bp + 8 * c);
    }
  };

  v8f acc[2][4];
  const v8f vzero = {0.f, 0.f, 0.f, 0.f, 0.f, 0.f, 0.f, 0.f};
#pragma unroll
  for (int mi = 0; mi < 2; ++mi)
#pragma unroll
    for (int ni = 0; ni < 4; ++ni) acc[mi][ni] = vzero;

  constexpr int KB = F / BK;  // 64
  issue(0, 0);
  for (int kb = 0; kb < KB; ++kb) {
    const int cur = kb & 1;
    wait_async0();
    __syncthreads();
    if (kb + 1 < KB) issue(cur ^ 1, kb + 1);
    const __bf16* Acur = smem + cur * BUFE;
    const __bf16* Bcur = Acur + BM * LDT;
#pragma unroll
    for (int ks = 0; ks < BK; ks += 32) {
      v16bf afr[2], bfr[4];
#pragma unroll
      for (int mi = 0; mi < 2; ++mi) afr[mi] = load_fragA(Acur, wm * 32 + mi * 16 + lm, ks, lane);
#pragma unroll
      for (int ni = 0; ni < 4; ++ni) bfr[ni] = load_fragB(Bcur, wn * 64 + ni * 16 + lm, ks, lane);
#pragma unroll
      for (int mi = 0; mi < 2; ++mi)
#pragma unroll
        for (int ni = 0; ni < 4; ++ni)
          acc[mi][ni] = __builtin_amdgcn_wmma_f32_16x16x32_bf16(
              false, afr[mi], false, bfr[ni], (short)0, acc[mi][ni], false, false);
    }
  }

  const int rOff = (lane < 16) ? 0 : 8;
#pragma unroll
  for (int ni = 0; ni < 4; ++ni) {
    int col = nBase + wn * 64 + ni * 16 + lm;
    float bias = b2[e * D + col];
#pragma unroll
    for (int mi = 0; mi < 2; ++mi) {
#pragma unroll
      for (int r = 0; r < 8; ++r) {
        int row = cBase + wm * 32 + mi * 16 + r + rOff;
        ye[((size_t)e * C + row) * D + col] = acc[mi][ni][r] + bias;
      }
    }
  }
}

// ---------------- kernel 7: gate-weighted combine ----------------
__global__ __launch_bounds__(256) void combine_kernel(
    const float* __restrict__ ye, const int* __restrict__ tok_slot,
    const float* __restrict__ topv, float* __restrict__ out) {
  size_t gid = (size_t)blockIdx.x * blockDim.x + threadIdx.x;
  int t = (int)(gid / (D / 4));
  int dq = (int)(gid % (D / 4));
  int s0 = tok_slot[t * 2], s1 = tok_slot[t * 2 + 1];
  float v0 = topv[t * 2], v1 = topv[t * 2 + 1];
  float4 acc = {0.f, 0.f, 0.f, 0.f};
  if (s0 >= 0) {
    float4 y = ((const float4*)ye)[(size_t)s0 * (D / 4) + dq];
    acc.x += v0 * y.x; acc.y += v0 * y.y; acc.z += v0 * y.z; acc.w += v0 * y.w;
  }
  if (s1 >= 0) {
    float4 y = ((const float4*)ye)[(size_t)s1 * (D / 4) + dq];
    acc.x += v1 * y.x; acc.y += v1 * y.y; acc.z += v1 * y.z; acc.w += v1 * y.w;
  }
  ((float4*)out)[gid] = acc;
}

// ---------------- host launcher ----------------
extern "C" void kernel_launch(void* const* d_in, const int* in_sizes, int n_in,
                              void* d_out, int out_size, void* d_ws, size_t ws_size,
                              hipStream_t stream) {
  const float* x  = (const float*)d_in[0];
  const float* gw = (const float*)d_in[1];
  const float* w1 = (const float*)d_in[2];
  const float* b1 = (const float*)d_in[3];
  const float* w2 = (const float*)d_in[4];
  const float* b2 = (const float*)d_in[5];
  float* out = (float*)d_out;

  char* ws = (char*)d_ws;
  size_t off = 0;
  auto alloc = [&](size_t bytes) -> void* {
    void* p = ws + off;
    off += (bytes + 255) & ~(size_t)255;
    return p;
  };
  __bf16* xb  = (__bf16*)alloc((size_t)T * D * 2);
  __bf16* w1t = (__bf16*)alloc((size_t)E * F * D * 2);   // [e][f][d]
  __bf16* w2t = (__bf16*)alloc((size_t)E * D * F * 2);   // [e][d][f]
  __bf16* hbuf = (__bf16*)alloc((size_t)E * C * F * 2);  // [e][c][f]
  float*  ye  = (float*)alloc((size_t)E * C * D * 4);    // [e][c][d]
  int*   eid  = (int*)alloc((size_t)T * K * 4);
  float* topv = (float*)alloc((size_t)T * K * 4);
  int*   tok_slot = (int*)alloc((size_t)T * K * 4);
  int*   slot_token = (int*)alloc((size_t)E * C * 4);

  gate_topk_kernel<<<T / 256, 256, 0, stream>>>(x, gw, eid, topv);
  dispatch_kernel<<<1, 512, 0, stream>>>(eid, tok_slot, slot_token);
  cvt_x_kernel<<<(T * D / 4) / 256, 256, 0, stream>>>(x, xb);
  dim3 tb(32, 8);
  tconv_kernel<<<dim3(F / 32, D / 32, E), tb, 0, stream>>>(w1, w1t, D, F);
  tconv_kernel<<<dim3(D / 32, F / 32, E), tb, 0, stream>>>(w2, w2t, F, D);
  gemm1_kernel<<<dim3(F / 128, C / 128, E), 256, 0, stream>>>(xb, w1t, b1, slot_token, hbuf);
  gemm2_kernel<<<dim3(D / 128, C / 128, E), 256, 0, stream>>>(hbuf, w2t, b2, ye);
  combine_kernel<<<(T * (D / 4)) / 256, 256, 0, stream>>>(ye, tok_slot, topv, out);
}
